// DistGATConv_35261681500222
// MI455X (gfx1250) — matrix operand order repaired and checked by reading, not compiled, not run on hardware
//
#include <hip/hip_runtime.h>
#include <math.h>

typedef __attribute__((ext_vector_type(16))) _Float16 v16h;
typedef __attribute__((ext_vector_type(8)))  _Float16 v8h;
typedef __attribute__((ext_vector_type(4)))  _Float16 v4h;
typedef __attribute__((ext_vector_type(8)))  float    v8f;

#define N_SRC 200000
#define N_DST 50000
#define NEDGE 800000
#define KIN   256
#define NH    3
#define FOUT  64
#define HF    192          /* NH*FOUT */
#define SLOPE 0.2f
#define CHUNK 96

// ---------------------------------------------------------------------------
// f32 -> f16 conversion, 4 elements per thread, grid-stride
// ---------------------------------------------------------------------------
__global__ void gat_cvt_f16(const float* __restrict__ in, _Float16* __restrict__ out, int n4) {
  int i = blockIdx.x * blockDim.x + threadIdx.x;
  int stride = gridDim.x * blockDim.x;
  for (int g = i; g < n4; g += stride) {
    float4 v = reinterpret_cast<const float4*>(in)[g];
    v4h h;
    h[0] = (_Float16)v.x; h[1] = (_Float16)v.y;
    h[2] = (_Float16)v.z; h[3] = (_Float16)v.w;
    reinterpret_cast<v4h*>(out)[g] = h;
  }
}

// ---------------------------------------------------------------------------
// feat = x @ W^T  via v_wmma_f32_16x16x32_f16.
// 128 threads = 4 waves; each wave computes one 16-row strip x all 192 cols.
// A fragment: lane holds row (lane&15), K = kk + (lane>>4)*16 + [0..15]
//             (one contiguous v16h load).
// B fragment: lane holds W-row ct*16+(lane&15), K = kk + (lane>>4)*16 + [0..15]
//             (one contiguous v16h load; B[k][n] = W[n][k]).
// C layout:   VGPR v, lane l -> row = v + 8*(l>>4), col = l&15.
// ---------------------------------------------------------------------------
__global__ __launch_bounds__(128) void gat_gemm_wmma(
    const _Float16* __restrict__ xh, const _Float16* __restrict__ wh,
    float* __restrict__ feat) {
  const int lane    = threadIdx.x & 31;
  const int wave    = threadIdx.x >> 5;
  const int rowTile = blockIdx.x * 4 + wave;        // 0..12499
  const int m       = lane & 15;
  const int hsel    = lane >> 4;                    // 0 or 1

  const _Float16* xrow = xh + (size_t)(rowTile * 16 + m) * KIN;

  v8f acc[12];
  const v8f vzero = {0.f, 0.f, 0.f, 0.f, 0.f, 0.f, 0.f, 0.f};
#pragma unroll
  for (int t = 0; t < 12; ++t) acc[t] = vzero;

  for (int kk = 0; kk < KIN; kk += 32) {
    v16h a = *reinterpret_cast<const v16h*>(xrow + kk + hsel * 16);
#pragma unroll
    for (int ct = 0; ct < 12; ++ct) {
      const _Float16* wrow = wh + (size_t)(ct * 16 + m) * KIN + kk + hsel * 16;
      v16h b = *reinterpret_cast<const v16h*>(wrow);
      acc[ct] = __builtin_amdgcn_wmma_f32_16x16x32_f16(
          false, a, false, b, (short)0, acc[ct], false, false);
    }
  }

#pragma unroll
  for (int ct = 0; ct < 12; ++ct) {
#pragma unroll
    for (int v = 0; v < 8; ++v) {
      int row = rowTile * 16 + hsel * 8 + v;
      feat[(size_t)row * HF + ct * 16 + m] = acc[ct][v];
    }
  }
}

// ---------------------------------------------------------------------------
// Per-node attention logits: el[i,h] = <feat[i,h,:], attn_l[h,:]>,
// er[i,h] likewise for i < N_DST.  One wave32 per node; lane covers 2 of the
// 64 features per head, butterfly reduce.
// ---------------------------------------------------------------------------
__global__ __launch_bounds__(256) void gat_logits(
    const float* __restrict__ feat, const float* __restrict__ al,
    const float* __restrict__ ar, float* __restrict__ el,
    float* __restrict__ er) {
  const int lane = threadIdx.x & 31;
  const int gw   = blockIdx.x * (blockDim.x >> 5) + (threadIdx.x >> 5);
  const int nw   = gridDim.x * (blockDim.x >> 5);
  for (int i = gw; i < N_SRC; i += nw) {
    const float* f = feat + (size_t)i * HF;
    const bool isDst = (i < N_DST);
    float pl[NH], pr[NH];
#pragma unroll
    for (int h = 0; h < NH; ++h) {
      float v0 = f[h * FOUT + lane];
      float v1 = f[h * FOUT + 32 + lane];
      pl[h] = v0 * al[h * FOUT + lane] + v1 * al[h * FOUT + 32 + lane];
      pr[h] = isDst ? (v0 * ar[h * FOUT + lane] + v1 * ar[h * FOUT + 32 + lane]) : 0.f;
    }
#pragma unroll
    for (int off = 16; off > 0; off >>= 1) {
#pragma unroll
      for (int h = 0; h < NH; ++h) {
        pl[h] += __shfl_xor(pl[h], off, 32);
        pr[h] += __shfl_xor(pr[h], off, 32);
      }
    }
    if (lane == 0) {
#pragma unroll
      for (int h = 0; h < NH; ++h) {
        el[(size_t)i * NH + h] = pl[h];
        if (isDst) er[(size_t)i * NH + h] = pr[h];
      }
    }
  }
}

// ---------------------------------------------------------------------------
// Fused edge softmax + weighted aggregation. dst is sorted, so one workgroup
// per dst node finds its edge segment by binary search; wave 0 computes the
// per-head max and exp-sum; then per-edge weights are staged in LDS in chunks
// and 192 threads (one per output channel) accumulate coalesced feature rows.
// No atomics -> deterministic.
// ---------------------------------------------------------------------------
__device__ __forceinline__ int gat_lbound(const int* __restrict__ a, int n, int v) {
  int lo = 0, hi = n;
  while (lo < hi) { int mid = (lo + hi) >> 1; if (a[mid] < v) lo = mid + 1; else hi = mid; }
  return lo;
}

__device__ __forceinline__ float gat_leaky(float x) { return x > 0.f ? x : SLOPE * x; }

__global__ __launch_bounds__(192) void gat_aggregate(
    const float* __restrict__ feat, const float* __restrict__ el,
    const float* __restrict__ er, const int* __restrict__ src,
    const int* __restrict__ dst, float* __restrict__ out) {
  __shared__ int   seg[2];
  __shared__ float erd[NH];
  __shared__ float mh[NH];
  __shared__ float invs[NH];
  __shared__ float wch[CHUNK * NH];

  const int d = blockIdx.x;
  const int tid = threadIdx.x;

  if (tid < 2) seg[tid] = gat_lbound(dst, NEDGE, d + tid);
  if (tid >= 32 && tid < 32 + NH) erd[tid - 32] = er[(size_t)d * NH + (tid - 32)];
  __syncthreads();

  const int lo = seg[0], hi = seg[1];

  if (tid < 32) {
    const float r0 = erd[0], r1 = erd[1], r2 = erd[2];
    float m0 = -3.0e38f, m1 = -3.0e38f, m2 = -3.0e38f;
    for (int e = lo + tid; e < hi; e += 32) {
      const int s = src[e];
      m0 = fmaxf(m0, gat_leaky(el[(size_t)s * NH + 0] + r0));
      m1 = fmaxf(m1, gat_leaky(el[(size_t)s * NH + 1] + r1));
      m2 = fmaxf(m2, gat_leaky(el[(size_t)s * NH + 2] + r2));
    }
#pragma unroll
    for (int off = 16; off > 0; off >>= 1) {
      m0 = fmaxf(m0, __shfl_xor(m0, off, 32));
      m1 = fmaxf(m1, __shfl_xor(m1, off, 32));
      m2 = fmaxf(m2, __shfl_xor(m2, off, 32));
    }
    if (hi == lo) { m0 = 0.f; m1 = 0.f; m2 = 0.f; }   // empty segment: ref -> 0
    float s0 = 0.f, s1 = 0.f, s2 = 0.f;
    for (int e = lo + tid; e < hi; e += 32) {
      const int s = src[e];
      s0 += __expf(gat_leaky(el[(size_t)s * NH + 0] + r0) - m0);
      s1 += __expf(gat_leaky(el[(size_t)s * NH + 1] + r1) - m1);
      s2 += __expf(gat_leaky(el[(size_t)s * NH + 2] + r2) - m2);
    }
#pragma unroll
    for (int off = 16; off > 0; off >>= 1) {
      s0 += __shfl_xor(s0, off, 32);
      s1 += __shfl_xor(s1, off, 32);
      s2 += __shfl_xor(s2, off, 32);
    }
    if (tid == 0) {
      mh[0] = m0; mh[1] = m1; mh[2] = m2;
      invs[0] = s0 > 0.f ? 1.f / s0 : 1.f;
      invs[1] = s1 > 0.f ? 1.f / s1 : 1.f;
      invs[2] = s2 > 0.f ? 1.f / s2 : 1.f;
    }
  }
  __syncthreads();

  const int h = tid / FOUT;     // head
  const int c = tid;            // output channel 0..191
  float accv = 0.f;

  for (int base = lo; base < hi; base += CHUNK) {
    const int n = min(CHUNK, hi - base);
    for (int t = tid; t < n * NH; t += 192) {
      const int e  = base + t / NH;
      const int hh = t - (t / NH) * NH;
      const int s  = src[e];
      const float a = gat_leaky(el[(size_t)s * NH + hh] + erd[hh]);
      wch[t] = __expf(a - mh[hh]) * invs[hh];
    }
    __syncthreads();
    for (int j = 0; j < n; ++j) {
      const int s = src[base + j];
      accv += wch[j * NH + h] * feat[(size_t)s * HF + c];
    }
    __syncthreads();
  }

  out[(size_t)d * HF + c] = accv;
}

// ---------------------------------------------------------------------------
// Launch
// ---------------------------------------------------------------------------
extern "C" void kernel_launch(void* const* d_in, const int* in_sizes, int n_in,
                              void* d_out, int out_size, void* d_ws, size_t ws_size,
                              hipStream_t stream) {
  const float* x      = (const float*)d_in[0];   // [N_SRC, KIN]
  const float* W      = (const float*)d_in[1];   // [HF, KIN]
  const float* attn_l = (const float*)d_in[2];   // [1, NH, FOUT]
  const float* attn_r = (const float*)d_in[3];   // [1, NH, FOUT]
  const int*   src    = (const int*)d_in[4];     // [NEDGE]
  const int*   dst    = (const int*)d_in[5];     // [NEDGE] sorted
  float*       out    = (float*)d_out;           // [N_DST, HF]

  // workspace layout (all offsets 256B aligned)
  char* ws = (char*)d_ws;
  const size_t XH_B   = (size_t)N_SRC * KIN * sizeof(_Float16);   // 102,400,000
  const size_t WH_B   = (size_t)HF * KIN * sizeof(_Float16);      //      98,304
  const size_t FEAT_B = (size_t)N_SRC * HF * sizeof(float);       // 153,600,000
  const size_t EL_B   = (size_t)N_SRC * NH * sizeof(float);       //   2,400,000
  _Float16* xh   = (_Float16*)(ws);
  _Float16* wh   = (_Float16*)(ws + XH_B);
  float*    feat = (float*)(ws + XH_B + WH_B);
  float*    el   = (float*)(ws + XH_B + WH_B + FEAT_B);
  float*    er   = (float*)(ws + XH_B + WH_B + FEAT_B + EL_B);

  (void)in_sizes; (void)n_in; (void)out_size; (void)ws_size;

  // 1) convert x and W to f16
  {
    int n4 = (N_SRC * KIN) / 4;                 // 12,800,000
    gat_cvt_f16<<<(n4 + 1023) / 1024, 256, 0, stream>>>(x, xh, n4);
    int w4 = (HF * KIN) / 4;                    // 12,288
    gat_cvt_f16<<<(w4 + 255) / 256, 256, 0, stream>>>(W, wh, w4);
  }

  // 2) feat = x @ W^T   (12500 row tiles, 4 per block)
  gat_gemm_wmma<<<12500 / 4, 128, 0, stream>>>(xh, wh, feat);

  // 3) attention logits (one wave per node; 8 waves/block)
  gat_logits<<<(N_SRC + 7) / 8, 256, 0, stream>>>(feat, attn_l, attn_r, el, er);

  // 4) fused softmax + aggregation, one workgroup per dst node
  gat_aggregate<<<N_DST, 192, 0, stream>>>(feat, el, er, src, dst, out);
}
